// RoFormerSelfAttention_63264868270198
// MI455X (gfx1250) — compile-verified, matrix operand order checked by reading, not run on hardware
//
#include <hip/hip_runtime.h>

// ---------------------------------------------------------------------------
// RoFormer self-attention for MI455X (gfx1250).
// B=2, S=2048, D=1024, H=16, HD=64.
// Compute-bound (~60 GFLOP vs ~100MB of traffic, all L2-resident): use the
// bf16 WMMA path (v_wmma_f32_16x16x32_bf16) with f32 accumulation, and the
// CDNA5 async global->LDS path (ASYNCcnt) to stage K/V tiles shared by the
// 8 waves of each attention workgroup.
// ---------------------------------------------------------------------------

#define Bsz 2
#define Ssz 2048
#define Dsz 1024
#define Hh  16
#define HDm 64

typedef __attribute__((ext_vector_type(16))) __bf16 v16bf;
typedef __attribute__((ext_vector_type(8)))  float  v8f;

struct BF16x16Bits { uint4 lo, hi; };

// 16 contiguous bf16 (32 bytes) -> A/B fragment register block.
// Works for global pointers (global_load_b128 x2) and LDS pointers
// (ds_load_b128 x2).
__device__ __forceinline__ v16bf load_bf16x16(const __bf16* p) {
    BF16x16Bits b;
    b.lo = *reinterpret_cast<const uint4*>(p);
    b.hi = *reinterpret_cast<const uint4*>(p + 8);
    return __builtin_bit_cast(v16bf, b);
}

__device__ __forceinline__ v8f wmma_bf16(v16bf a, v16bf b, v8f c) {
    // D = A(16x32 bf16) * B(32x16 bf16) + C(16x16 f32)
    return __builtin_amdgcn_wmma_f32_16x16x32_bf16(
        /*neg_a=*/false, a, /*neg_b=*/false, b,
        /*c_mod=*/(short)0, c, /*reuse_a=*/false, /*reuse_b=*/false);
}

// Async global->LDS 16-byte copy (CDNA5, tracked by ASYNCcnt).
// lds_off: LDS byte address (low 32 bits of a generic pointer to __shared__).
__device__ __forceinline__ void async_cp_b128(unsigned lds_off, const __bf16* g) {
    asm volatile("global_load_async_to_lds_b128 %0, %1, off"
                 :: "v"(lds_off), "v"(g) : "memory");
}

__device__ __forceinline__ unsigned lds_addr(const void* p) {
    return (unsigned)(uintptr_t)p;   // generic LDS ptr: addr[31:0] = LDS offset
}

// ---------------------------------------------------------------------------
// Prep kernel 1: f32 -> bf16, same layout (for X, row-major M x K).
// ---------------------------------------------------------------------------
__global__ void cvt_f32_to_bf16(const float* __restrict__ in,
                                __bf16* __restrict__ out, int n4) {
    int i = blockIdx.x * blockDim.x + threadIdx.x;
    if (i < n4) {
        float4 v = reinterpret_cast<const float4*>(in)[i];
        __bf16 o[4] = { (__bf16)v.x, (__bf16)v.y, (__bf16)v.z, (__bf16)v.w };
        *reinterpret_cast<uint2*>(out + (size_t)i * 4) =
            *reinterpret_cast<uint2*>(o);
    }
}

// ---------------------------------------------------------------------------
// Prep kernel 2: W (K x N, f32) -> Wt (N x K, bf16) so B-fragments (one
// column of W per lane) become contiguous 32B loads.
// ---------------------------------------------------------------------------
__global__ void transpose_w_bf16(const float* __restrict__ W,
                                 __bf16* __restrict__ Wt) {
    int tid = blockIdx.x * blockDim.x + threadIdx.x;   // 0 .. 1M-1
    int k = tid >> 10;          // row of W  (coalesced read along n)
    int n = tid & (Dsz - 1);
    Wt[(size_t)n * Dsz + k] = (__bf16)W[(size_t)k * Dsz + n];
}

// ---------------------------------------------------------------------------
// Kernel 3: fused QKV projection + RoPE.
// One wave per 16x16 output tile; shares the A fragment across the three
// weight matrices (3 WMMAs per k-step). RoPE pairs adjacent columns, which
// live in adjacent lanes of the C fragment -> shfl_xor(.,1).
// Outputs: Q,K as (B,H,S,HD) bf16; V transposed as (B,H,HD,S) bf16.
// ---------------------------------------------------------------------------
__global__ void qkv_rope_wmma(const __bf16* __restrict__ X,
                              const __bf16* __restrict__ Wqt,
                              const __bf16* __restrict__ Wkt,
                              const __bf16* __restrict__ Wvt,
                              const float*  __restrict__ sinT,  // (S, HD/2)
                              const float*  __restrict__ cosT,
                              __bf16* __restrict__ Qo,
                              __bf16* __restrict__ Ko,
                              __bf16* __restrict__ Vt) {
    int wave  = blockIdx.x * (blockDim.x >> 5) + (threadIdx.x >> 5);
    int lane  = threadIdx.x & 31;
    int lrow  = lane & 15;
    int lhalf = lane >> 4;
    int mt = wave >> 6;              // 0..255 : 16-row strip of the 4096 rows
    int nt = wave & 63;              // 0..63  : 16-col strip of the 1024 cols
    int m0 = mt << 4, n0 = nt << 4;

    const __bf16* aP = X + (size_t)(m0 + lrow) * Dsz + lhalf * 16;
    int n_col = n0 + lrow;           // column this lane owns in B fragments
    const __bf16* bq = Wqt + (size_t)n_col * Dsz + lhalf * 16;
    const __bf16* bk = Wkt + (size_t)n_col * Dsz + lhalf * 16;
    const __bf16* bv = Wvt + (size_t)n_col * Dsz + lhalf * 16;

    v8f cq = {}, ck = {}, cv = {};
#pragma unroll 4
    for (int kk = 0; kk < Dsz; kk += 32) {
        v16bf a = load_bf16x16(aP + kk);
        cq = wmma_bf16(a, load_bf16x16(bq + kk), cq);
        ck = wmma_bf16(a, load_bf16x16(bk + kk), ck);
        cv = wmma_bf16(a, load_bf16x16(bv + kk), cv);
    }

    // --- RoPE + stores --------------------------------------------------
    int bb = m0 >> 11;               // batch (tiles never straddle batches)
    int h  = n_col >> 6;
    int d  = n_col & (HDm - 1);
    int i2 = d >> 1;
    float sgn = (n_col & 1) ? 1.f : -1.f;   // even dim: x*cos - y*sin

    unsigned short vpack[8];
#pragma unroll
    for (int g = 0; g < 8; ++g) {
        int row = m0 + g + 8 * lhalf;
        int s   = row & (Ssz - 1);
        float sv = sinT[s * (HDm / 2) + i2];
        float cw = cosT[s * (HDm / 2) + i2];
        float xq = cq[g], xk = ck[g], xv = cv[g];
        float yq = __shfl_xor(xq, 1, 32);
        float yk = __shfl_xor(xk, 1, 32);
        float yv = __shfl_xor(xv, 1, 32);
        float rq = xq * cw + sgn * yq * sv;
        float rk = xk * cw + sgn * yk * sv;
        float rv = xv * cw + sgn * yv * sv;
        size_t qi = ((size_t)(bb * Hh + h) * Ssz + s) * HDm + d;
        Qo[qi] = (__bf16)rq;
        Ko[qi] = (__bf16)rk;
        vpack[g] = __builtin_bit_cast(unsigned short, (__bf16)rv);
    }
    // V transposed: row = head-dim d, cols = 8 consecutive positions.
    size_t vb = ((size_t)(bb * Hh + h) * HDm + d) * Ssz
              + (m0 & (Ssz - 1)) + 8 * lhalf;
    *reinterpret_cast<uint4*>(Vt + vb) = *reinterpret_cast<uint4*>(vpack);
}

// ---------------------------------------------------------------------------
// Kernel 4: flash attention. 8 waves/block, all on the SAME (b,h); each wave
// owns one 16-query strip. K/V^T 32-key tiles are staged into double-buffered
// LDS with async global->LDS copies (ASYNCcnt) so all 8 waves share one copy.
//
// Math trick: compute S^T = K*Q^T so each lane owns one query's score column
// (softmax = 15 in-lane maxes + one shfl_xor(16)), and P^T's C-layout matches
// the A-layout for P*V after one half-wave shfl_xor(16).
// ---------------------------------------------------------------------------
__global__ __launch_bounds__(256) void attn_wmma(const __bf16* __restrict__ Q,
                                                 const __bf16* __restrict__ K,
                                                 const __bf16* __restrict__ Vt,
                                                 float* __restrict__ out) {
    // double-buffered tiles: K 32x64 bf16 (4KB), V^T 64x32 bf16 (4KB)
    __shared__ __attribute__((aligned(16))) __bf16 lsK[2][32 * HDm];
    __shared__ __attribute__((aligned(16))) __bf16 lsV[2][HDm * 32];

    int tid   = threadIdx.x;
    int wave  = tid >> 5;
    int lane  = tid & 31;
    int lrow  = lane & 15;
    int lhalf = lane >> 4;

    int bh = blockIdx.x >> 4;        // 0..31  (b*16+h)
    int qg = blockIdx.x & 15;        // q-group: 8 strips of 16 queries
    int bb = bh >> 4, h = bh & 15;
    int q0 = (qg * 8 + wave) << 4;

    const __bf16* Qb = Q  + (size_t)bh * Ssz * HDm;
    const __bf16* Kb = K  + (size_t)bh * Ssz * HDm;
    const __bf16* Vb = Vt + (size_t)bh * HDm * Ssz;

    // Q^T as B-fragment: lane owns query column q0+lrow; HD split 2x32.
    v16bf qb0 = load_bf16x16(Qb + (size_t)(q0 + lrow) * HDm + lhalf * 16);
    v16bf qb1 = load_bf16x16(Qb + (size_t)(q0 + lrow) * HDm + lhalf * 16 + 32);

    // Per-thread staging slots: K uses 256 threads x 16B (32 rows x 128B),
    // V uses 256 threads x 16B (64 rows x 64B).
    int kRow = tid >> 3, kSeg = tid & 7;   // K: row, 8-elem segment
    int vRow = tid >> 2, vSeg = tid & 3;   // V: row, 8-elem segment

    // Prologue: stage chunk 0 into buffer 0 (2 async instr per wave).
    async_cp_b128(lds_addr(&lsK[0][kRow * HDm + kSeg * 8]),
                  Kb + (size_t)kRow * HDm + kSeg * 8);
    async_cp_b128(lds_addr(&lsV[0][vRow * 32 + vSeg * 8]),
                  Vb + (size_t)vRow * Ssz + vSeg * 8);

    float mrun = -1e30f, lrun = 0.f;
    v8f o0 = {}, o1 = {}, o2 = {}, o3 = {};
    const float sc = 0.125f;         // 1/sqrt(64)

    for (int k0 = 0; k0 < Ssz; k0 += 32) {
        int p = (k0 >> 5) & 1;
        if (k0 + 32 < Ssz) {
            // stage next chunk into the other buffer, then wait only for
            // the current chunk's 2 transfers (leave the new 2 in flight)
            async_cp_b128(lds_addr(&lsK[p ^ 1][kRow * HDm + kSeg * 8]),
                          Kb + (size_t)(k0 + 32 + kRow) * HDm + kSeg * 8);
            async_cp_b128(lds_addr(&lsV[p ^ 1][vRow * 32 + vSeg * 8]),
                          Vb + (size_t)vRow * Ssz + (k0 + 32) + vSeg * 8);
            asm volatile("s_wait_asynccnt 0x2" ::: "memory");
        } else {
            asm volatile("s_wait_asynccnt 0x0" ::: "memory");
        }
        __syncthreads();             // all waves' transfers landed in LDS

        const __bf16* lk = lsK[p];
        const __bf16* lv = lsV[p];

        v8f st0 = {}, st1 = {};
        st0 = wmma_bf16(load_bf16x16(lk + lrow * HDm + lhalf * 16),        qb0, st0);
        st0 = wmma_bf16(load_bf16x16(lk + lrow * HDm + lhalf * 16 + 32),   qb1, st0);
        st1 = wmma_bf16(load_bf16x16(lk + (16 + lrow) * HDm + lhalf * 16), qb0, st1);
        st1 = wmma_bf16(load_bf16x16(lk + (16 + lrow) * HDm + lhalf * 16 + 32), qb1, st1);

        // Per-lane: 16 scores, all for query q0+lrow (keys split across the
        // half-wave partner lane).
        float s0[8], s1[8];
        float mx = -1e30f;
#pragma unroll
        for (int g = 0; g < 8; ++g) {
            s0[g] = st0[g] * sc;
            s1[g] = st1[g] * sc;
            mx = fmaxf(mx, fmaxf(s0[g], s1[g]));
        }
        mx = fmaxf(mx, __shfl_xor(mx, 16, 32));
        float mnew = fmaxf(mrun, mx);
        float corr = __expf(mrun - mnew);
        float ssum = 0.f;
#pragma unroll
        for (int g = 0; g < 8; ++g) {
            s0[g] = __expf(s0[g] - mnew);
            s1[g] = __expf(s1[g] - mnew);
            ssum += s0[g] + s1[g];
        }
        ssum += __shfl_xor(ssum, 16, 32);
        lrun = lrun * corr + ssum;
        mrun = mnew;

        // Rescale O (rows q = g + 8*lhalf; stats live in lane q) and build
        // the P A-fragment (16q x 32keys) from the P^T C-fragments.
        v16bf pa;
#pragma unroll
        for (int g = 0; g < 8; ++g) {
            float cg = __shfl(corr, g + 8 * lhalf, 32);
            o0[g] *= cg; o1[g] *= cg; o2[g] *= cg; o3[g] *= cg;
            float t0 = __shfl_xor(s0[g], 16, 32);
            float t1 = __shfl_xor(s1[g], 16, 32);
            float lo = lhalf ? t1 : s0[g];   // keys base+g
            float hi = lhalf ? s1[g] : t0;   // keys base+8+g
            pa[g]     = (__bf16)lo;
            pa[g + 8] = (__bf16)hi;
        }

        // O += P * V   (V^T rows are head-dims -> contiguous B fragments)
        o0 = wmma_bf16(pa, load_bf16x16(lv + ( 0 + lrow) * 32 + lhalf * 16), o0);
        o1 = wmma_bf16(pa, load_bf16x16(lv + (16 + lrow) * 32 + lhalf * 16), o1);
        o2 = wmma_bf16(pa, load_bf16x16(lv + (32 + lrow) * 32 + lhalf * 16), o2);
        o3 = wmma_bf16(pa, load_bf16x16(lv + (48 + lrow) * 32 + lhalf * 16), o3);

        __syncthreads();             // reads done before buffer is re-staged
    }

    float linv = 1.f / lrun;
#pragma unroll
    for (int g = 0; g < 8; ++g) {
        float lg = __shfl(linv, g + 8 * lhalf, 32);
        int row = q0 + g + 8 * lhalf;
        size_t base = ((size_t)bb * Ssz + row) * Dsz + h * HDm;
        out[base +  0 + lrow] = o0[g] * lg;
        out[base + 16 + lrow] = o1[g] * lg;
        out[base + 32 + lrow] = o2[g] * lg;
        out[base + 48 + lrow] = o3[g] * lg;
    }
}

// ---------------------------------------------------------------------------
// Launch. Workspace layout (bytes):
//   Xbf  @ 0         (8 MB)   X in bf16, row-major
//   Wqt  @ 8388608   (2 MB)   Wq^T bf16
//   Wkt  @ 10485760  (2 MB)
//   Wvt  @ 12582912  (2 MB)
//   Qd   @ 14680064  (8 MB)   (B,H,S,HD) bf16 (post-RoPE)
//   Kd   @ 23068672  (8 MB)
//   Vtd  @ 31457280  (8 MB)   (B,H,HD,S) bf16 (post-RoPE, transposed)
// total 39,845,888 bytes.
// ---------------------------------------------------------------------------
extern "C" void kernel_launch(void* const* d_in, const int* in_sizes, int n_in,
                              void* d_out, int out_size, void* d_ws, size_t ws_size,
                              hipStream_t stream) {
    const float* hs   = (const float*)d_in[0];
    const float* Wq   = (const float*)d_in[1];
    const float* Wk   = (const float*)d_in[2];
    const float* Wv   = (const float*)d_in[3];
    const float* sinT = (const float*)d_in[4];
    const float* cosT = (const float*)d_in[5];
    float* out = (float*)d_out;

    char* w = (char*)d_ws;
    __bf16* Xbf = (__bf16*)(w + 0);
    __bf16* Wqt = (__bf16*)(w + 8388608);
    __bf16* Wkt = (__bf16*)(w + 10485760);
    __bf16* Wvt = (__bf16*)(w + 12582912);
    __bf16* Qd  = (__bf16*)(w + 14680064);
    __bf16* Kd  = (__bf16*)(w + 23068672);
    __bf16* Vtd = (__bf16*)(w + 31457280);

    // X: 4096*1024 = 4M elems -> 1M float4s
    cvt_f32_to_bf16<<<4096, 256, 0, stream>>>(hs, Xbf, 1048576);
    transpose_w_bf16<<<4096, 256, 0, stream>>>(Wq, Wqt);
    transpose_w_bf16<<<4096, 256, 0, stream>>>(Wk, Wkt);
    transpose_w_bf16<<<4096, 256, 0, stream>>>(Wv, Wvt);

    // 256 m-tiles * 64 n-tiles = 16384 waves, 8 waves/block
    qkv_rope_wmma<<<2048, 256, 0, stream>>>(Xbf, Wqt, Wkt, Wvt, sinT, cosT,
                                            Qd, Kd, Vtd);
    // 32 (b,h) * 16 q-groups = 512 blocks, 8 waves each on one (b,h)
    attn_wmma<<<512, 256, 0, stream>>>(Qd, Kd, Vtd, out);
}
